// TopoGNN_60919816127110
// MI455X (gfx1250) — matrix operand order repaired and checked by reading, not compiled, not run on hardware
//
#include <hip/hip_runtime.h>
#include <math.h>

// ---------------------------------------------------------------------------
// TopoGNN on MI455X (gfx1250): WMMA f16 (f32 accum) for edge MLP, node update.
// Edge kernel: weights cached in LDS (320KB WGP), M=32 edges per wave.
// ---------------------------------------------------------------------------

#define N_NODES 100000
#define N_EDGES 800000
#define N_GRAPH 64
#define F_NODE  91
#define F_EDGE  20
#define HDIM    128
#define WAVES_PER_BLOCK 4   // node_update
#define EDGE_WAVES 8        // edge_msg

// LDS geometry for edge_msg (dwords)
#define W1_DW   18432       // 9*8 tiles * 256 dwords  (288x128 f16)
#define W2_DW   8192        // 4*8 tiles * 256 dwords  (128x128 f16)
#define A_DW    4608        // 32 rows x 144 dwords    (32x288 f16)
#define HID_DW  2048        // 32 rows x 64 dwords     (32x128 f16)
#define SLICE_DW (A_DW + HID_DW)
#define EDGE_LDS_BYTES ((W1_DW + W2_DW + EDGE_WAVES * SLICE_DW) * 4)  // 319,488

typedef __attribute__((ext_vector_type(16))) _Float16 v16h;
typedef __attribute__((ext_vector_type(8)))  float    v8f;
typedef __attribute__((ext_vector_type(4)))  unsigned v4u;

union V16H { v4u q[2]; v16h h; };

static __device__ __forceinline__ unsigned pack_f16x2(float a, float b) {
  _Float16 ha = (_Float16)a, hb = (_Float16)b;
  unsigned short ua = __builtin_bit_cast(unsigned short, ha);
  unsigned short ub = __builtin_bit_cast(unsigned short, hb);
  return (unsigned)ua | ((unsigned)ub << 16);
}

// ---------------- workspace layout (bytes) ----------------
static const size_t OFF_H    = 0;                    // [N,128] f32
static const size_t OFF_H16  = 51200000;             // [N,128] f16 (packed dwords)
static const size_t OFF_AGG  = 76800000;             // [N,128] f32
static const size_t OFF_W1F  = 128000000;            // [3][9][8][32][8] dwords (fused 288x128 f16, B layout)
static const size_t OFF_W2F  = 128221184;            // [3][4][8][32][8]
static const size_t OFF_WUF  = 128319488;            // [3][8][8][32][8]
static const size_t OFF_B1F  = 128516096;            // [3][128] f32 (fused bias)
static const size_t OFF_PSUM = 128517632;            // [64][128] f32
static const size_t OFF_PMAX = 128550400;            // [64][128] f32
static const size_t OFF_CNT  = 128583168;            // [64] i32

// ===========================================================================
// Weight prep: convert to f16 in this code's WMMA B lane layout (contiguous k, as A uses).
//  B tile (32K x 16N): lane<16 -> K=0..15 col=lane ; lane>=16 -> K=16..31.
//  VGPR j holds K pair (2j, 2j+1). Stored as [tile][lane][j] dwords so a lane
//  fetches its 8 dwords with two contiguous b128 loads.
// ===========================================================================
__global__ void prep_msg_w1(const float* __restrict__ msg_W1,
                            const float* __restrict__ edge_W,
                            unsigned* __restrict__ W1f) {
  int kt = blockIdx.x, nt = blockIdx.y, l = blockIdx.z;   // 9 x 8 x 3
  int t = threadIdx.x, lane = t & 31, j = t >> 5;
  int n = nt * 16 + (lane & 15);
  int kbase = kt * 32 + ((lane & 16) ? 16 : 0) + 2 * j;
  float v[2];
  for (int h2 = 0; h2 < 2; ++h2) {
    int k = kbase + h2;
    float x = 0.f;
    if (k < 2 * HDIM) {
      x = msg_W1[((size_t)l * 384 + k) * HDIM + n];
    } else {
      int k2 = k - 2 * HDIM;
      if (k2 < F_EDGE) {              // fused: edge_W @ msg_W1[256:,:]
        float s = 0.f;
        for (int tt = 0; tt < HDIM; ++tt)
          s += edge_W[k2 * HDIM + tt] * msg_W1[((size_t)l * 384 + 256 + tt) * HDIM + n];
        x = s;
      }
    }
    v[h2] = x;
  }
  W1f[((((size_t)l * 9 + kt) * 8 + nt) * 32 + lane) * 8 + j] = pack_f16x2(v[0], v[1]);
}

__global__ void prep_plain_w(const float* __restrict__ W, unsigned* __restrict__ Wf,
                             int KT, int Krows) {
  int kt = blockIdx.x, nt = blockIdx.y, l = blockIdx.z;
  int t = threadIdx.x, lane = t & 31, j = t >> 5;
  int n = nt * 16 + (lane & 15);
  int kbase = kt * 32 + ((lane & 16) ? 16 : 0) + 2 * j;
  float a = (kbase     < Krows) ? W[((size_t)l * Krows + kbase)     * HDIM + n] : 0.f;
  float b = (kbase + 1 < Krows) ? W[((size_t)l * Krows + kbase + 1) * HDIM + n] : 0.f;
  Wf[((((size_t)l * KT + kt) * 8 + nt) * 32 + lane) * 8 + j] = pack_f16x2(a, b);
}

__global__ void prep_b1(const float* __restrict__ msg_W1, const float* __restrict__ msg_b1,
                        const float* __restrict__ edge_b, float* __restrict__ b1f) {
  int l = blockIdx.x, n = threadIdx.x;
  float s = msg_b1[l * HDIM + n];
  for (int t = 0; t < HDIM; ++t)
    s += edge_b[t] * msg_W1[((size_t)l * 384 + 256 + t) * HDIM + n];
  b1f[l * HDIM + n] = s;
}

// ===========================================================================
// Node embedding: h = node_feats @ node_W + node_b  (tiny: VALU)
// ===========================================================================
__global__ void node_embed(const float* __restrict__ nf, const float* __restrict__ W,
                           const float* __restrict__ b, float* __restrict__ h,
                           unsigned* __restrict__ h16) {
  __shared__ float row[F_NODE];
  __shared__ float out[HDIM];
  int n = blockIdx.x, j = threadIdx.x;     // 128 threads
  if (j < F_NODE) row[j] = nf[(size_t)n * F_NODE + j];
  __syncthreads();
  float s = b[j];
  for (int k = 0; k < F_NODE; ++k) s += row[k] * W[k * HDIM + j];
  h[(size_t)n * HDIM + j] = s;
  out[j] = s;
  __syncthreads();
  if ((j & 1) == 0) h16[(size_t)n * 64 + (j >> 1)] = pack_f16x2(out[j], out[j + 1]);
}

// ===========================================================================
// Edge message kernel: per-wave 32-edge tile (2 x 16-row A tiles share B).
//  Weights for the active layer staged once per block into LDS.
//  layer1: [32 x 288] @ [288 x 128] -> relu -> layer2: [32x128]@[128x128]
//  scatter-add (nan_to_num) into agg[dst].
// ===========================================================================
__global__ void __launch_bounds__(EDGE_WAVES * 32)
edge_msg(const int* __restrict__ edge_index,
         const float* __restrict__ edge_feats,
         const unsigned* __restrict__ h16,
         const unsigned* __restrict__ W1f,
         const unsigned* __restrict__ W2f,
         const float* __restrict__ b1f,
         const float* __restrict__ msg_b2,
         float* __restrict__ agg, int layer) {
  extern __shared__ char smem[];
  unsigned* Wlds  = (unsigned*)smem;            // W1: [0, W1_DW)
  unsigned* W2lds = Wlds + W1_DW;               // W2: [W1_DW, W1_DW+W2_DW)
  const int lane = threadIdx.x & 31;
  const int wave = threadIdx.x >> 5;
  unsigned* A   = Wlds + (W1_DW + W2_DW) + (size_t)wave * SLICE_DW; // 32x288 f16
  unsigned* Hid = A + A_DW;                                         // 32x128 f16

  // ---- block-cooperative weight staging (once; L2 -> LDS)
  const unsigned* W1L = W1f + (size_t)layer * W1_DW;
  const unsigned* W2L = W2f + (size_t)layer * W2_DW;
  for (int i = threadIdx.x; i < W1_DW; i += EDGE_WAVES * 32) Wlds[i] = W1L[i];
  for (int i = threadIdx.x; i < W2_DW; i += EDGE_WAVES * 32) W2lds[i] = W2L[i];
  __syncthreads();

  const int* src = edge_index;
  const int* dst = edge_index + N_EDGES;

  const int numTiles = (N_EDGES + 31) / 32;
  const int row   = lane & 15;
  const int khalf = (lane >> 4) & 1;
  const int rb    = khalf ? 8 : 0;

  for (int tile = blockIdx.x * EDGE_WAVES + wave; tile < numTiles;
       tile += gridDim.x * EDGE_WAVES) {
    int e0 = tile * 32;
    // ---- stage A panel: 32 rows of [h_src(128) | h_dst(128) | ef(20 pad 32)] f16
    for (int r = 0; r < 32; ++r) {
      int e = e0 + r; if (e >= N_EDGES) e = N_EDGES - 1;
      int s = src[e], d = dst[e];
      const unsigned* hs = h16 + (size_t)s * 64;
      const unsigned* hd = h16 + (size_t)d * 64;
      unsigned* arow = A + r * 144;
      arow[lane]           = hs[lane];
      arow[lane + 32]      = hs[lane + 32];
      arow[64 + lane]      = hd[lane];
      arow[64 + lane + 32] = hd[lane + 32];
      if (lane < 16) {
        unsigned v = 0;
        if (lane < 10) {
          const float* ef = edge_feats + (size_t)e * F_EDGE + lane * 2;
          v = pack_f16x2(ef[0], ef[1]);
        }
        arow[128 + lane] = v;
      }
    }
    __builtin_amdgcn_wave_barrier();
    asm volatile("s_wait_dscnt 0" ::: "memory");

    // ---- layer 1: K=288 (9 chunks of 32), N=128 (8 col tiles), M=32 (2 tiles)
    v8f acc0[8], acc1[8];
    #pragma unroll
    for (int nc = 0; nc < 8; ++nc) {
      float bb = b1f[layer * HDIM + nc * 16 + row];
      v8f z = {bb, bb, bb, bb, bb, bb, bb, bb};
      acc0[nc] = z; acc1[nc] = z;
    }
    #pragma unroll 1
    for (int kc = 0; kc < 9; ++kc) {
      const unsigned* ar0 = A + row * 144        + kc * 16 + khalf * 8;
      const unsigned* ar1 = A + (16 + row) * 144 + kc * 16 + khalf * 8;
      V16H a0, a1;
      a0.q[0] = *(const v4u*)ar0; a0.q[1] = *(const v4u*)(ar0 + 4);
      a1.q[0] = *(const v4u*)ar1; a1.q[1] = *(const v4u*)(ar1 + 4);
      #pragma unroll
      for (int nc = 0; nc < 8; ++nc) {
        const unsigned* bp = Wlds + (((size_t)kc * 8 + nc) * 32 + lane) * 8;
        V16H b; b.q[0] = *(const v4u*)bp; b.q[1] = *(const v4u*)(bp + 4);
        acc0[nc] = __builtin_amdgcn_wmma_f32_16x16x32_f16(
            false, a0.h, false, b.h, (short)0, acc0[nc], false, false);
        acc1[nc] = __builtin_amdgcn_wmma_f32_16x16x32_f16(
            false, a1.h, false, b.h, (short)0, acc1[nc], false, false);
      }
    }
    // relu -> f16 hidden panel in LDS (32 x 128)
    #pragma unroll
    for (int nc = 0; nc < 8; ++nc) {
      int col = nc * 16 + row;
      #pragma unroll
      for (int r = 0; r < 8; ++r) {
        float v0 = acc0[nc][r]; v0 = v0 > 0.f ? v0 : 0.f;
        float v1 = acc1[nc][r]; v1 = v1 > 0.f ? v1 : 0.f;
        ((_Float16*)Hid)[(rb + r) * 128 + col]      = (_Float16)v0;
        ((_Float16*)Hid)[(16 + rb + r) * 128 + col] = (_Float16)v1;
      }
    }
    __builtin_amdgcn_wave_barrier();
    asm volatile("s_wait_dscnt 0" ::: "memory");

    // ---- layer 2: K=128 (4 chunks), N=128, M=32
    #pragma unroll
    for (int nc = 0; nc < 8; ++nc) {
      float bb = msg_b2[layer * HDIM + nc * 16 + row];
      v8f z = {bb, bb, bb, bb, bb, bb, bb, bb};
      acc0[nc] = z; acc1[nc] = z;
    }
    #pragma unroll 1
    for (int kc = 0; kc < 4; ++kc) {
      const unsigned* ar0 = Hid + row * 64        + kc * 16 + khalf * 8;
      const unsigned* ar1 = Hid + (16 + row) * 64 + kc * 16 + khalf * 8;
      V16H a0, a1;
      a0.q[0] = *(const v4u*)ar0; a0.q[1] = *(const v4u*)(ar0 + 4);
      a1.q[0] = *(const v4u*)ar1; a1.q[1] = *(const v4u*)(ar1 + 4);
      #pragma unroll
      for (int nc = 0; nc < 8; ++nc) {
        const unsigned* bp = W2lds + (((size_t)kc * 8 + nc) * 32 + lane) * 8;
        V16H b; b.q[0] = *(const v4u*)bp; b.q[1] = *(const v4u*)(bp + 4);
        acc0[nc] = __builtin_amdgcn_wmma_f32_16x16x32_f16(
            false, a0.h, false, b.h, (short)0, acc0[nc], false, false);
        acc1[nc] = __builtin_amdgcn_wmma_f32_16x16x32_f16(
            false, a1.h, false, b.h, (short)0, acc1[nc], false, false);
      }
    }
    // ---- nan_to_num + scatter-add to agg[dst] (L2-resident atomics)
    #pragma unroll
    for (int r = 0; r < 8; ++r) {
      int e = e0 + rb + r;
      if (e < N_EDGES) {
        int d = dst[e];
        #pragma unroll
        for (int nc = 0; nc < 8; ++nc) {
          float v = acc0[nc][r];
          if (!__builtin_isfinite(v)) v = 0.f;
          atomicAdd(&agg[(size_t)d * HDIM + nc * 16 + row], v);
        }
      }
      int e2 = e0 + 16 + rb + r;
      if (e2 < N_EDGES) {
        int d = dst[e2];
        #pragma unroll
        for (int nc = 0; nc < 8; ++nc) {
          float v = acc1[nc][r];
          if (!__builtin_isfinite(v)) v = 0.f;
          atomicAdd(&agg[(size_t)d * HDIM + nc * 16 + row], v);
        }
      }
    }
  }
}

// ===========================================================================
// Node update: h_new = relu([h|agg] @ upd_W + b); h = LN(h + h_new).
// Per-wave 16-node tile; WMMA K=256; fused residual + layernorm.
// ===========================================================================
__global__ void node_update(float* __restrict__ h, unsigned* __restrict__ h16,
                            const float* __restrict__ agg,
                            const unsigned* __restrict__ WuF,
                            const float* __restrict__ upd_b,
                            const float* __restrict__ ln_g,
                            const float* __restrict__ ln_b, int layer) {
  extern __shared__ char smem[];
  const int lane = threadIdx.x & 31;
  const int wave = threadIdx.x >> 5;
  unsigned* A = (unsigned*)smem + (size_t)wave * (2048 + 2048); // 16x256 f16
  float*    Y = (float*)(A + 2048);                             // 16x128 f32
  const unsigned* WuL = WuF + (size_t)layer * 8 * 8 * 32 * 8;
  const float* lg = ln_g + layer * HDIM;
  const float* lb = ln_b + layer * HDIM;

  const int numTiles = (N_NODES + 15) / 16;
  const int row   = lane & 15;
  const int khalf = (lane >> 4) & 1;
  const int rb    = khalf ? 8 : 0;

  for (int tile = blockIdx.x * WAVES_PER_BLOCK + wave; tile < numTiles;
       tile += gridDim.x * WAVES_PER_BLOCK) {
    int n0 = tile * 16;
    for (int r = 0; r < 16; ++r) {
      int n = n0 + r; if (n >= N_NODES) n = N_NODES - 1;
      unsigned* arow = A + r * 128;
      arow[lane]      = h16[(size_t)n * 64 + lane];
      arow[lane + 32] = h16[(size_t)n * 64 + lane + 32];
      const float* ag = agg + (size_t)n * HDIM;
      arow[64 + lane] = pack_f16x2(ag[lane * 2],      ag[lane * 2 + 1]);
      arow[96 + lane] = pack_f16x2(ag[64 + lane * 2], ag[64 + lane * 2 + 1]);
    }
    __builtin_amdgcn_wave_barrier();
    asm volatile("s_wait_dscnt 0" ::: "memory");

    v8f acc[8];
    #pragma unroll
    for (int nc = 0; nc < 8; ++nc) {
      float bb = upd_b[layer * HDIM + nc * 16 + row];
      v8f z = {bb, bb, bb, bb, bb, bb, bb, bb};
      acc[nc] = z;
    }
    #pragma unroll 1
    for (int kc = 0; kc < 8; ++kc) {
      const unsigned* arow = A + row * 128 + kc * 16 + khalf * 8;
      V16H a; a.q[0] = *(const v4u*)arow; a.q[1] = *(const v4u*)(arow + 4);
      #pragma unroll
      for (int nc = 0; nc < 8; ++nc) {
        const unsigned* bp = WuL + (((size_t)kc * 8 + nc) * 32 + lane) * 8;
        V16H b; b.q[0] = *(const v4u*)bp; b.q[1] = *(const v4u*)(bp + 4);
        acc[nc] = __builtin_amdgcn_wmma_f32_16x16x32_f16(
            false, a.h, false, b.h, (short)0, acc[nc], false, false);
      }
    }
    // relu(h_new) -> Y (f32)
    #pragma unroll
    for (int nc = 0; nc < 8; ++nc) {
      int col = nc * 16 + row;
      #pragma unroll
      for (int r = 0; r < 8; ++r) {
        float v = acc[nc][r];
        Y[(rb + r) * 128 + col] = v > 0.f ? v : 0.f;
      }
    }
    __builtin_amdgcn_wave_barrier();
    asm volatile("s_wait_dscnt 0" ::: "memory");

    // residual + layernorm: lanes 0..15 each own one row
    if (lane < 16) {
      int n = n0 + lane;
      if (n < N_NODES) {
        float* yr = Y + lane * 128;
        float s = 0.f, s2 = 0.f;
        for (int j = 0; j < HDIM; ++j) {
          float y = h[(size_t)n * HDIM + j] + yr[j];
          yr[j] = y;
          s += y; s2 += y * y;
        }
        float mu  = s  * (1.f / HDIM);
        float var = s2 * (1.f / HDIM) - mu * mu;
        float rs  = rsqrtf(var + 1e-5f);
        for (int j = 0; j < HDIM; j += 2) {
          float v0 = (yr[j]     - mu) * rs * lg[j]     + lb[j];
          float v1 = (yr[j + 1] - mu) * rs * lg[j + 1] + lb[j + 1];
          h[(size_t)n * HDIM + j]     = v0;
          h[(size_t)n * HDIM + j + 1] = v1;
          h16[(size_t)n * 64 + (j >> 1)] = pack_f16x2(v0, v1);
        }
      }
    }
    __builtin_amdgcn_wave_barrier();
    asm volatile("s_wait_dscnt 0" ::: "memory");
  }
}

// ===========================================================================
// Pooling + readout
// ===========================================================================
__global__ void pool_init(float* psum, float* pmax, int* cnt) {
  int i = blockIdx.x * blockDim.x + threadIdx.x;
  if (i < N_GRAPH * HDIM) { psum[i] = 0.f; pmax[i] = -__builtin_inff(); }
  if (i < N_GRAPH) cnt[i] = 0;
}

__global__ void pool_scatter(const float* __restrict__ h, const int* __restrict__ batch,
                             float* __restrict__ psum, float* __restrict__ pmax,
                             int* __restrict__ cnt) {
  size_t i = (size_t)blockIdx.x * blockDim.x + threadIdx.x;
  if (i >= (size_t)N_NODES * HDIM) return;
  int n = (int)(i >> 7), j = (int)(i & 127);
  int g = batch[n];
  float v = h[i];
  atomicAdd(&psum[g * HDIM + j], v);
  unsigned* ua = (unsigned*)&pmax[g * HDIM + j];
  unsigned cur = *ua;
  while (__uint_as_float(cur) < v) {
    unsigned prev = atomicCAS(ua, cur, __float_as_uint(v));
    if (prev == cur) break;
    cur = prev;
  }
  if (j == 0) atomicAdd(&cnt[g], 1);
}

__global__ void readout(const float* __restrict__ psum, const float* __restrict__ pmax,
                        const int* __restrict__ cnt,
                        const float* __restrict__ W1, const float* __restrict__ b1,
                        const float* __restrict__ W2, const float* __restrict__ b2,
                        const float* __restrict__ W3, const float* __restrict__ b3,
                        float* __restrict__ out) {
  __shared__ float hg[2 * HDIM];
  __shared__ float x1[HDIM];
  __shared__ float x2[HDIM / 2];
  int g = blockIdx.x, t = threadIdx.x;     // 128 threads
  float c = (float)cnt[g]; if (c < 1.f) c = 1.f;
  float mean = psum[g * HDIM + t] / c;
  float mx = pmax[g * HDIM + t];
  if (!__builtin_isfinite(mx)) mx = 0.f;
  hg[t] = mean; hg[HDIM + t] = mx;
  __syncthreads();
  float s = b1[t];
  for (int k = 0; k < 2 * HDIM; ++k) s += hg[k] * W1[k * HDIM + t];
  x1[t] = s > 0.f ? s : 0.f;
  __syncthreads();
  if (t < HDIM / 2) {
    float s2 = b2[t];
    for (int k = 0; k < HDIM; ++k) s2 += x1[k] * W2[k * (HDIM / 2) + t];
    x2[t] = s2 > 0.f ? s2 : 0.f;
  }
  __syncthreads();
  if (t == 0) {
    float s3 = b3[0];
    for (int k = 0; k < HDIM / 2; ++k) s3 += x2[k] * W3[k];
    out[g] = s3;
  }
}

// ===========================================================================
extern "C" void kernel_launch(void* const* d_in, const int* in_sizes, int n_in,
                              void* d_out, int out_size, void* d_ws, size_t ws_size,
                              hipStream_t stream) {
  (void)in_sizes; (void)n_in; (void)out_size; (void)ws_size;
  const float* node_feats = (const float*)d_in[0];
  const int*   edge_index = (const int*)  d_in[1];
  const float* edge_feats = (const float*)d_in[2];
  const int*   batch      = (const int*)  d_in[3];
  const float* node_W = (const float*)d_in[4];
  const float* node_b = (const float*)d_in[5];
  const float* edge_W = (const float*)d_in[6];
  const float* edge_b = (const float*)d_in[7];
  const float* msg_W1 = (const float*)d_in[8];
  const float* msg_b1 = (const float*)d_in[9];
  const float* msg_W2 = (const float*)d_in[10];
  const float* msg_b2 = (const float*)d_in[11];
  const float* upd_W  = (const float*)d_in[12];
  const float* upd_b  = (const float*)d_in[13];
  const float* ln_g   = (const float*)d_in[14];
  const float* ln_b   = (const float*)d_in[15];
  const float* ro_W1  = (const float*)d_in[16];
  const float* ro_b1  = (const float*)d_in[17];
  const float* ro_W2  = (const float*)d_in[18];
  const float* ro_b2  = (const float*)d_in[19];
  const float* ro_W3  = (const float*)d_in[20];
  const float* ro_b3  = (const float*)d_in[21];

  char* ws = (char*)d_ws;
  float*    h    = (float*)   (ws + OFF_H);
  unsigned* h16  = (unsigned*)(ws + OFF_H16);
  float*    agg  = (float*)   (ws + OFF_AGG);
  unsigned* W1f  = (unsigned*)(ws + OFF_W1F);
  unsigned* W2f  = (unsigned*)(ws + OFF_W2F);
  unsigned* WuF  = (unsigned*)(ws + OFF_WUF);
  float*    b1f  = (float*)   (ws + OFF_B1F);
  float*    psum = (float*)   (ws + OFF_PSUM);
  float*    pmax = (float*)   (ws + OFF_PMAX);
  int*      cnt  = (int*)     (ws + OFF_CNT);

  // weight prep (tiny)
  prep_msg_w1 <<<dim3(9, 8, 3), 256, 0, stream>>>(msg_W1, edge_W, W1f);
  prep_plain_w<<<dim3(4, 8, 3), 256, 0, stream>>>(msg_W2, W2f, 4, HDIM);
  prep_plain_w<<<dim3(8, 8, 3), 256, 0, stream>>>(upd_W,  WuF, 8, 2 * HDIM);
  prep_b1     <<<3, 128, 0, stream>>>(msg_W1, msg_b1, edge_b, b1f);

  node_embed<<<N_NODES, 128, 0, stream>>>(node_feats, node_W, node_b, h, h16);

  for (int l = 0; l < 3; ++l) {
    hipMemsetAsync(agg, 0, (size_t)N_NODES * HDIM * sizeof(float), stream);
    edge_msg<<<512, EDGE_WAVES * 32, EDGE_LDS_BYTES, stream>>>(
        edge_index, edge_feats, h16, W1f, W2f, b1f, msg_b2, agg, l);
    node_update<<<800, WAVES_PER_BLOCK * 32, WAVES_PER_BLOCK * (2048 + 2048) * 4, stream>>>(
        h, h16, agg, WuF, upd_b, ln_g, ln_b, l);
  }

  pool_init<<<(N_GRAPH * HDIM + 255) / 256, 256, 0, stream>>>(psum, pmax, cnt);
  pool_scatter<<<(int)(((size_t)N_NODES * HDIM + 255) / 256), 256, 0, stream>>>(
      h, batch, psum, pmax, cnt);
  readout<<<N_GRAPH, 128, 0, stream>>>(psum, pmax, cnt, ro_W1, ro_b1, ro_W2, ro_b2,
                                       ro_W3, ro_b3, (float*)d_out);
}
